// CoAttention_ImageDNS_13391708029218
// MI455X (gfx1250) — matrix operand order repaired and checked by reading, not compiled, not run on hardware
//
#include <hip/hip_runtime.h>
#include <math.h>

// CDNA5 / gfx1250 wave32 WMMA types
typedef __attribute__((ext_vector_type(16))) __bf16 v16bf;
typedef __attribute__((ext_vector_type(8)))  __bf16 v8bf;
typedef __attribute__((ext_vector_type(4)))  __bf16 bf16x4;
typedef __attribute__((ext_vector_type(8)))  float  v8f;

#define HDIM 1024
#define SDIM 512
#define BDIM 32
#define RDIM 196

__device__ __forceinline__ __bf16 bf16_trunc(float f) {
    unsigned short h = (unsigned short)(__float_as_uint(f) >> 16);
    return __builtin_bit_cast(__bf16, h);
}
__device__ __forceinline__ float bf16_hi_f32(float f) {
    return __uint_as_float(__float_as_uint(f) & 0xffff0000u);
}
// branch-free tanh: no exec-mask games, ~1e-6 accurate
__device__ __forceinline__ float fast_tanh(float x) {
    float xc = fminf(fmaxf(x, -15.f), 15.f);
    float e  = __expf(2.f * xc);
    return (e - 1.f) * __builtin_amdgcn_rcpf(e + 1.f);
}

// Split an f32 array into bf16 hi/lo planes (residual-correction pair).
__global__ __launch_bounds__(256)
void split_bf16_kernel(const float* __restrict__ src,
                       __bf16* __restrict__ hi, __bf16* __restrict__ lo, int n4)
{
    int idx = blockIdx.x * 256 + threadIdx.x;
    if (idx >= n4) return;
    float4 f = ((const float4*)src)[idx];
    bf16x4 h, l;
    h[0] = bf16_trunc(f.x); l[0] = bf16_trunc(f.x - bf16_hi_f32(f.x));
    h[1] = bf16_trunc(f.y); l[1] = bf16_trunc(f.y - bf16_hi_f32(f.y));
    h[2] = bf16_trunc(f.z); l[2] = bf16_trunc(f.z - bf16_hi_f32(f.z));
    h[3] = bf16_trunc(f.w); l[3] = bf16_trunc(f.w - bf16_hi_f32(f.w));
    ((bf16x4*)hi)[idx] = h;
    ((bf16x4*)lo)[idx] = l;
}

// t[b, row] = sum_o wvec[o] * tanh( sum_h X[b,row,h] * W[o,h] )
// One workgroup (8 wave32) per (16-row M-tile, batch).
// A-tile pre-split into bf16 hi/lo LDS planes; W pre-split in global workspace.
__global__ __launch_bounds__(256)
void score_gemm_kernel(const float* __restrict__ X,
                       const __bf16* __restrict__ Wh, const __bf16* __restrict__ Wl,
                       const float* __restrict__ wvec, float* __restrict__ t,
                       int Rb)
{
    __shared__ __bf16 AhS[16 * HDIM];   // 32KB (first 16 floats reused as sRow)
    __shared__ __bf16 AlS[16 * HDIM];   // 32KB
    const int b     = blockIdx.y;
    const int mtile = blockIdx.x;
    const int tid   = threadIdx.x;
    const int wave  = tid >> 5;
    const int lane  = tid & 31;
    const int half  = lane >> 4;   // lane group 0..15 / 16..31
    const int m     = lane & 15;   // A-row index, and B/C column index N

    const float* Xb = X + (size_t)b * Rb * HDIM;

    // Cooperative A-tile load + bf16 hi/lo split (16 x 1024), rows clamped.
    #pragma unroll
    for (int j = 0; j < 16; ++j) {
        int idx = (tid + j * 256) * 4;
        int r   = idx >> 10;          // / HDIM
        int k   = idx & (HDIM - 1);
        int rg  = mtile * 16 + r;
        if (rg >= Rb) rg = Rb - 1;
        float4 f = *(const float4*)&Xb[(size_t)rg * HDIM + k];
        bf16x4 h, l;
        h[0] = bf16_trunc(f.x); l[0] = bf16_trunc(f.x - bf16_hi_f32(f.x));
        h[1] = bf16_trunc(f.y); l[1] = bf16_trunc(f.y - bf16_hi_f32(f.y));
        h[2] = bf16_trunc(f.z); l[2] = bf16_trunc(f.z - bf16_hi_f32(f.z));
        h[3] = bf16_trunc(f.w); l[3] = bf16_trunc(f.w - bf16_hi_f32(f.w));
        *(bf16x4*)&AhS[idx] = h;
        *(bf16x4*)&AlS[idx] = l;
    }
    __syncthreads();

    float rAcc[8];
    #pragma unroll
    for (int v = 0; v < 8; ++v) rAcc[v] = 0.f;

    const __bf16* ArowH = &AhS[m * HDIM];
    const __bf16* ArowL = &AlS[m * HDIM];

    // Each wave handles N-tiles wave, wave+8, ... (64 tiles of 16 columns)
    for (int nt = wave; nt < 64; nt += 8) {
        const int    nbase = nt * 16;
        const float  wv    = wvec[nbase + m];  // column weight for this lane's N
        const __bf16* BrowH = Wh + (size_t)(nbase + m) * HDIM;
        const __bf16* BrowL = Wl + (size_t)(nbase + m) * HDIM;

        v8f acc = {};
        for (int kc = 0; kc < HDIM; kc += 32) {
            // A fragment (16x32 bf16 layout): lane covers k = kc+half*8+[0..7]
            // and kc+16+half*8+[0..7]; two 16B ds_load_b128 per plane.
            v8bf a0h = *(const v8bf*)&ArowH[kc + half * 8];
            v8bf a1h = *(const v8bf*)&ArowH[kc + 16 + half * 8];
            v8bf a0l = *(const v8bf*)&ArowL[kc + half * 8];
            v8bf a1l = *(const v8bf*)&ArowL[kc + 16 + half * 8];
            // B fragment (32x16 bf16 layout): lane covers k = kc+half*8+[0..7]
            // and kc+16+half*8+[0..7]; two 16B global_load_b128 per plane.
            v8bf b0h = *(const v8bf*)&BrowH[kc + half * 8];
            v8bf b1h = *(const v8bf*)&BrowH[kc + 16 + half * 8];
            v8bf b0l = *(const v8bf*)&BrowL[kc + half * 8];
            v8bf b1l = *(const v8bf*)&BrowL[kc + 16 + half * 8];

            v16bf ah = __builtin_shufflevector(a0h, a1h, 0,1,2,3,4,5,6,7,8,9,10,11,12,13,14,15);
            v16bf al = __builtin_shufflevector(a0l, a1l, 0,1,2,3,4,5,6,7,8,9,10,11,12,13,14,15);
            v16bf bh = __builtin_shufflevector(b0h, b1h, 0,1,2,3,4,5,6,7,8,9,10,11,12,13,14,15);
            v16bf bl = __builtin_shufflevector(b0l, b1l, 0,1,2,3,4,5,6,7,8,9,10,11,12,13,14,15);

            // bf16x3: P = Ah*Bh + Al*Bh + Ah*Bl  (f32 accumulate) ~ fp32 exact
            acc = __builtin_amdgcn_wmma_f32_16x16x32_bf16(false, ah, false, bh,
                                                          (short)0, acc, false, false);
            acc = __builtin_amdgcn_wmma_f32_16x16x32_bf16(false, al, false, bh,
                                                          (short)0, acc, false, false);
            acc = __builtin_amdgcn_wmma_f32_16x16x32_bf16(false, ah, false, bl,
                                                          (short)0, acc, false, false);
        }

        // Epilogue: tanh, scale by wvec[n], reduce over the 16 N-lanes.
        // C layout: VGPR v, lanes 0-15 -> row v; lanes 16-31 -> row v+8; N = lane&15.
        #pragma unroll
        for (int v = 0; v < 8; ++v) {
            float val = fast_tanh(acc[v]) * wv;
            #pragma unroll
            for (int off = 1; off < 16; off <<= 1)
                val += __shfl_xor(val, off, 32);
            rAcc[v] += val;
        }
    }

    // Merge the 8 waves' partial row sums (reuse AhS[0..31] as 16 floats).
    __syncthreads();
    float* sRow = (float*)AhS;
    if (tid < 16) sRow[tid] = 0.f;
    __syncthreads();
    if (lane == 0) {
        #pragma unroll
        for (int v = 0; v < 8; ++v) atomicAdd(&sRow[v], rAcc[v]);
    } else if (lane == 16) {
        #pragma unroll
        for (int v = 0; v < 8; ++v) atomicAdd(&sRow[v + 8], rAcc[v]);
    }
    __syncthreads();
    if (tid < 16) {
        int rg = mtile * 16 + tid;
        if (rg < Rb) t[(size_t)b * Rb + rg] = sRow[tid];
    }
}

// In-place softmax over each batch's score row (n <= 512).
__global__ __launch_bounds__(256)
void softmax_rows_kernel(float* __restrict__ t, int n)
{
    __shared__ float red[256];
    const int b   = blockIdx.x;
    const int tid = threadIdx.x;
    float* row = t + (size_t)b * n;

    float mx = -INFINITY;
    for (int i = tid; i < n; i += 256) mx = fmaxf(mx, row[i]);
    red[tid] = mx; __syncthreads();
    for (int s = 128; s > 0; s >>= 1) {
        if (tid < s) red[tid] = fmaxf(red[tid], red[tid + s]);
        __syncthreads();
    }
    mx = red[0]; __syncthreads();

    float sm = 0.f;
    for (int i = tid; i < n; i += 256) {
        float e = expf(row[i] - mx);
        row[i] = e;
        sm += e;
    }
    red[tid] = sm; __syncthreads();
    for (int s = 128; s > 0; s >>= 1) {
        if (tid < s) red[tid] += red[tid + s];
        __syncthreads();
    }
    float inv = 1.f / red[0];
    for (int i = tid; i < n; i += 256) row[i] *= inv;
}

// out[b,s,h] = sum_r p[b,r] * X[b,r,h]   (result independent of s -> broadcast)
__global__ __launch_bounds__(256)
void attend_broadcast_kernel(const float* __restrict__ X, const float* __restrict__ p,
                             float* __restrict__ out, int Rb)
{
    __shared__ float sp[SDIM];
    const int b = blockIdx.y;
    const int h = blockIdx.x * 256 + threadIdx.x;

    for (int i = threadIdx.x; i < Rb; i += 256)
        sp[i] = p[(size_t)b * Rb + i];
    __syncthreads();

    const float* Xb = X + (size_t)b * Rb * HDIM;
    float acc = 0.f;
    for (int r = 0; r < Rb; ++r)
        acc += sp[r] * Xb[(size_t)r * HDIM + h];

    float* ob = out + (size_t)b * SDIM * HDIM + h;
    for (int s = 0; s < SDIM; ++s)
        ob[(size_t)s * HDIM] = acc;
}

extern "C" void kernel_launch(void* const* d_in, const int* in_sizes, int n_in,
                              void* d_out, int out_size, void* d_ws, size_t ws_size,
                              hipStream_t stream) {
    (void)in_sizes; (void)n_in; (void)out_size; (void)ws_size;
    const float* dns    = (const float*)d_in[0];   // (B,S,H)
    const float* img    = (const float*)d_in[1];   // (B,R,H)
    const float* W_i1   = (const float*)d_in[4];   // (H,H)
    const float* w_att1 = (const float*)d_in[5];   // (2H,)  -> wB = w_att1 + H
    const float* W_d2   = (const float*)d_in[7];   // (H,H)
    const float* w_att2 = (const float*)d_in[10];  // (2H,)  -> wD = w_att2 + H
    // W_d1, b_d1, wA, b_att1, W_i2, b_i2, wI, b_att2 cancel in the softmaxes.

    char* ws = (char*)d_ws;
    float*  t1  = (float*)ws;                            // B*R  = 6272 floats
    float*  t2  = t1 + BDIM * RDIM;                      // B*S  = 16384 floats
    __bf16* W1h = (__bf16*)(ws + 96 * 1024);             // 2MB each plane
    __bf16* W1l = W1h + (size_t)HDIM * HDIM;
    __bf16* W2h = W1l + (size_t)HDIM * HDIM;
    __bf16* W2l = W2h + (size_t)HDIM * HDIM;             // ws usage ~8.1MB total

    float* out_dns = (float*)d_out;                                   // (B,S,H)
    float* out_img = (float*)d_out + (size_t)BDIM * SDIM * HDIM;      // (B,S,H)

    dim3 blk(256);
    const int n4 = HDIM * HDIM / 4;          // 262144 float4 per weight matrix
    split_bf16_kernel<<<n4 / 256, blk, 0, stream>>>(W_i1, W1h, W1l, n4);
    split_bf16_kernel<<<n4 / 256, blk, 0, stream>>>(W_d2, W2h, W2l, n4);

    // Stage 1 score: t1[b,r] = wB . tanh(img[b,r,:] @ W_i1^T)
    score_gemm_kernel<<<dim3(13, BDIM), blk, 0, stream>>>(img, W1h, W1l, w_att1 + HDIM, t1, RDIM);
    // Stage 2 score: t2[b,j] = wD . tanh(dns[b,j,:] @ W_d2^T)
    score_gemm_kernel<<<dim3(32, BDIM), blk, 0, stream>>>(dns, W2h, W2l, w_att2 + HDIM, t2, SDIM);

    softmax_rows_kernel<<<BDIM, blk, 0, stream>>>(t1, RDIM);
    softmax_rows_kernel<<<BDIM, blk, 0, stream>>>(t2, SDIM);

    // att_img_features (second output): p1-weighted sum of img, broadcast over S
    attend_broadcast_kernel<<<dim3(HDIM / 256, BDIM), blk, 0, stream>>>(img, t1, out_img, RDIM);
    // att_dns_features (first output): p2-weighted sum of dns, broadcast over S
    attend_broadcast_kernel<<<dim3(HDIM / 256, BDIM), blk, 0, stream>>>(dns, t2, out_dns, SDIM);
}
